// net_crossInteraction_55533927137413
// MI455X (gfx1250) — matrix operand order repaired and checked by reading, not compiled, not run on hardware
//
#include <hip/hip_runtime.h>
#include <math.h>

typedef __attribute__((ext_vector_type(16))) _Float16 v16h;
typedef __attribute__((ext_vector_type(8)))  float    v8f;

__device__ __forceinline__ float actf(float v, int act) {
    if (act == 1)      return v > 0.f ? v : 0.f;
    else if (act == 2) return 1.f / (1.f + __expf(-v));
    else if (act == 3) return v > 0.f ? v : 0.1f * v;
    else if (act == 4) return tanhf(v);
    return v;
}

// ---------------------------------------------------------------------------
// Batched WMMA GEMM:  C[b] = post( act( A[b] @ op(B[b]) + bias ) )
//   A: (M,K) f32 row-major (lda % 4 == 0, 16B-aligned rows)
//   transB=1 -> B is (N,K) row-major (weights / activations^T), vectorized
//   transB=0 -> B is (K,N) row-major, strided scalar loads
// One wave computes a 16x32 C tile (2 WMMAs / K-chunk, shared A fragment).
// Guards are hoisted: load indices are clamped (safe), stores are guarded.
// ---------------------------------------------------------------------------
__global__ __launch_bounds__(128)
void wmma_gemm_kernel(const float* __restrict__ A, const float* __restrict__ Bw,
                      const float* __restrict__ bias, const float* __restrict__ addp,
                      float* __restrict__ C,
                      int M, int N, int K,
                      long long sA, long long sB, long long sC,
                      int lda, int ldb, int ldc,
                      int transB, int act, int reluA)
{
    const int wave  = threadIdx.x >> 5;
    const int lane  = threadIdx.x & 31;
    const int tileM = blockIdx.y << 4;
    const int tileN = blockIdx.x * 128 + wave * 32;
    if (tileN >= N) return;                        // wave-uniform exit
    const int b = blockIdx.z;

    const float* Ab = A  + (long long)b * sA;
    const float* Bb = Bw + (long long)b * sB;

    const int g   = lane >> 4;
    const int mr  = tileM + (lane & 15);
    const int nc0 = tileN + (lane & 15);
    const int nc1 = nc0 + 16;
    const int mL  = mr  < M ? mr  : M - 1;         // clamped (safe) load indices
    const int nL0 = nc0 < N ? nc0 : N - 1;
    const int nL1 = nc1 < N ? nc1 : N - 1;

    const float* Arow  = Ab + (long long)mL  * lda;
    const float* Brow0 = Bb + (long long)nL0 * ldb;   // transB path
    const float* Brow1 = Bb + (long long)nL1 * ldb;

    v8f acc0 = {0.f,0.f,0.f,0.f,0.f,0.f,0.f,0.f};
    v8f acc1 = {0.f,0.f,0.f,0.f,0.f,0.f,0.f,0.f};

    const int kFull = K & ~31;

#define CVA(x) ((_Float16)(reluA ? fmaxf((x), 0.f) : (x)))
    for (int k0 = 0; k0 < kFull; k0 += 32) {
        v16h af, bf0, bf1;
        // A fragment: run [k0+g*16, +16), contiguous b128s
        {
            const float4 a0 = *(const float4*)(Arow + k0 + g * 16);
            const float4 a1 = *(const float4*)(Arow + k0 + g * 16 + 4);
            const float4 a2 = *(const float4*)(Arow + k0 + g * 16 + 8);
            const float4 a3 = *(const float4*)(Arow + k0 + g * 16 + 12);
            af[0]=CVA(a0.x); af[1]=CVA(a0.y); af[2]=CVA(a0.z); af[3]=CVA(a0.w);
            af[4]=CVA(a1.x); af[5]=CVA(a1.y); af[6]=CVA(a1.z); af[7]=CVA(a1.w);
            af[8]=CVA(a2.x); af[9]=CVA(a2.y); af[10]=CVA(a2.z); af[11]=CVA(a2.w);
            af[12]=CVA(a3.x); af[13]=CVA(a3.y); af[14]=CVA(a3.z); af[15]=CVA(a3.w);
        }
        if (transB) {
            const float4 b0 = *(const float4*)(Brow0 + k0 + g * 16);
            const float4 b1 = *(const float4*)(Brow0 + k0 + g * 16 + 4);
            const float4 b2 = *(const float4*)(Brow0 + k0 + g * 16 + 8);
            const float4 b3 = *(const float4*)(Brow0 + k0 + g * 16 + 12);
            bf0[0]=(_Float16)b0.x; bf0[1]=(_Float16)b0.y; bf0[2]=(_Float16)b0.z; bf0[3]=(_Float16)b0.w;
            bf0[4]=(_Float16)b1.x; bf0[5]=(_Float16)b1.y; bf0[6]=(_Float16)b1.z; bf0[7]=(_Float16)b1.w;
            bf0[8]=(_Float16)b2.x; bf0[9]=(_Float16)b2.y; bf0[10]=(_Float16)b2.z; bf0[11]=(_Float16)b2.w;
            bf0[12]=(_Float16)b3.x; bf0[13]=(_Float16)b3.y; bf0[14]=(_Float16)b3.z; bf0[15]=(_Float16)b3.w;
            const float4 c0 = *(const float4*)(Brow1 + k0 + g * 16);
            const float4 c1 = *(const float4*)(Brow1 + k0 + g * 16 + 4);
            const float4 c2 = *(const float4*)(Brow1 + k0 + g * 16 + 8);
            const float4 c3 = *(const float4*)(Brow1 + k0 + g * 16 + 12);
            bf1[0]=(_Float16)c0.x; bf1[1]=(_Float16)c0.y; bf1[2]=(_Float16)c0.z; bf1[3]=(_Float16)c0.w;
            bf1[4]=(_Float16)c1.x; bf1[5]=(_Float16)c1.y; bf1[6]=(_Float16)c1.z; bf1[7]=(_Float16)c1.w;
            bf1[8]=(_Float16)c2.x; bf1[9]=(_Float16)c2.y; bf1[10]=(_Float16)c2.z; bf1[11]=(_Float16)c2.w;
            bf1[12]=(_Float16)c3.x; bf1[13]=(_Float16)c3.y; bf1[14]=(_Float16)c3.z; bf1[15]=(_Float16)c3.w;
        } else {
#pragma unroll
            for (int e = 0; e < 16; ++e) {
                const float* row = Bb + (long long)(k0 + g * 16 + e) * ldb;
                bf0[e] = (_Float16)row[nL0];
                bf1[e] = (_Float16)row[nL1];
            }
        }
        acc0 = __builtin_amdgcn_wmma_f32_16x16x32_f16(false, af, false, bf0,
                                                      (short)0, acc0, false, false);
        acc1 = __builtin_amdgcn_wmma_f32_16x16x32_f16(false, af, false, bf1,
                                                      (short)0, acc1, false, false);
    }

    if (kFull < K) {   // K remainder: clamped unconditional loads + value select
        const int k0 = kFull;
        v16h af, bf0, bf1;
#pragma unroll
        for (int e = 0; e < 8; ++e) {
            int k1 = k0 + g * 16 + e;
            float v1 = Arow[k1 < K ? k1 : K - 1];
            if (reluA) v1 = fmaxf(v1, 0.f);
            af[e] = (_Float16)(k1 < K ? v1 : 0.f);
            int k2 = k0 + g * 16 + 8 + e;
            float v2 = Arow[k2 < K ? k2 : K - 1];
            if (reluA) v2 = fmaxf(v2, 0.f);
            af[8 + e] = (_Float16)(k2 < K ? v2 : 0.f);
        }
#pragma unroll
        for (int e = 0; e < 16; ++e) {
            int k  = k0 + g * 16 + e;
            int kc = k < K ? k : K - 1;
            float u0, u1;
            if (transB) { u0 = Brow0[kc]; u1 = Brow1[kc]; }
            else { u0 = Bb[(long long)kc * ldb + nL0]; u1 = Bb[(long long)kc * ldb + nL1]; }
            bf0[e] = (_Float16)(k < K ? u0 : 0.f);
            bf1[e] = (_Float16)(k < K ? u1 : 0.f);
        }
        acc0 = __builtin_amdgcn_wmma_f32_16x16x32_f16(false, af, false, bf0,
                                                      (short)0, acc0, false, false);
        acc1 = __builtin_amdgcn_wmma_f32_16x16x32_f16(false, af, false, bf1,
                                                      (short)0, acc1, false, false);
    }
#undef CVA

    const float bv0 = bias ? bias[nL0] : 0.f;
    const float bv1 = bias ? bias[nL1] : 0.f;
    float* Cb = C + (long long)b * sC;
    const float* Rb = addp ? (addp + (long long)b * sC) : (const float*)0;
#pragma unroll
    for (int r = 0; r < 8; ++r) {
        int m = tileM + g * 8 + r;
        if (m < M) {
            long long rowoff = (long long)m * ldc;
            if (nc0 < N) {
                float v = actf(acc0[r] + bv0, act);
                if (Rb) v += Rb[rowoff + nc0];
                Cb[rowoff + nc0] = v;
            }
            if (nc1 < N) {
                float v = actf(acc1[r] + bv1, act);
                if (Rb) v += Rb[rowoff + nc1];
                Cb[rowoff + nc1] = v;
            }
        }
    }
}

// --------------------------- elementwise kernels ---------------------------

__global__ void zero_kernel(float* p, int n) {
    int i = blockIdx.x * blockDim.x + threadIdx.x;
    if (i < n) p[i] = 0.f;
}

// zero-pad rows: (rows, sw) -> (rows, dw), dw >= sw
__global__ void pad_copy_kernel(const float* __restrict__ src, float* __restrict__ dst,
                                int rows, int sw, int dw) {
    int i = blockIdx.x * blockDim.x + threadIdx.x;
    if (i >= rows * dw) return;
    int r = i / dw, c = i - r * dw;
    dst[i] = (c < sw) ? src[r * sw + c] : 0.f;
}

__global__ void embed_kernel(const int* __restrict__ ids,
                             const float* __restrict__ tab,
                             float* __restrict__ out, int rows) {
    int i = blockIdx.x * blockDim.x + threadIdx.x;
    if (i >= rows * 256) return;
    int r = i >> 8, d = i & 255;
    out[i] = tab[ids[r] * 256 + d];
}

__global__ void gru_step_kernel(const float* __restrict__ xp,
                                const float* __restrict__ gh,
                                float* __restrict__ h, float* __restrict__ ys,
                                int Nn, int T, int t) {
    int i = blockIdx.x * blockDim.x + threadIdx.x;
    if (i >= Nn * 256) return;
    int n = i >> 8, d = i & 255;
    long long xb = ((long long)n * T + t) * 768;
    long long gb = (long long)n * 768;
    float xr = xp[xb + d], xz = xp[xb + 256 + d], xn = xp[xb + 512 + d];
    float hr = gh[gb + d], hz = gh[gb + 256 + d], hn = gh[gb + 512 + d];
    float r  = 1.f / (1.f + __expf(-(xr + hr)));
    float z  = 1.f / (1.f + __expf(-(xz + hz)));
    float nn = tanhf(xn + r * hn);
    float hv = (1.f - z) * nn + z * h[i];
    h[i] = hv;
    ys[((long long)n * T + t) * 256 + d] = hv;
}

__global__ void gat_norm_kernel(float* __restrict__ a, const float* __restrict__ adj) {
    int b = blockIdx.y, row = blockIdx.x, tid = threadIdx.x;
    float* ar = a + ((long long)b * 1000 + row) * 1000;
    const float* jr = adj + ((long long)b * 1000 + row) * 1000;
    __shared__ float sb[256];
    float loc = 0.f;
    for (int c = tid; c < 1000; c += 256) {
        float adjv = (c == row) ? 1.f : jr[c];
        float v = (ar[c] + ((c == row) ? 1e-5f : 0.f)) * adjv;
        ar[c] = v;
        loc += v;
    }
    sb[tid] = loc; __syncthreads();
    for (int s = 128; s > 0; s >>= 1) {
        if (tid < s) sb[tid] += sb[tid + s];
        __syncthreads();
    }
    float inv = 1.f / sb[0];
    for (int c = tid; c < 1000; c += 256) ar[c] *= inv;
}

__global__ void ci_dot_kernel(const float* __restrict__ X, const float* __restrict__ Y,
                              float* __restrict__ out) {
    int bi = blockIdx.x, tid = threadIdx.x;
    __shared__ float sb[256];
    long long base = (long long)bi * 256;
    sb[tid] = X[base + tid] * Y[base + tid];
    __syncthreads();
    for (int s = 128; s > 0; s >>= 1) {
        if (tid < s) sb[tid] += sb[tid + s];
        __syncthreads();
    }
    if (tid == 0) out[bi] = tanhf(sb[0]) + 1.f;
}

__global__ void concat_kernel(const float* __restrict__ seq, const float* __restrict__ pg,
                              const float* __restrict__ ci0, const float* __restrict__ ci1,
                              float* __restrict__ u) {
    int i = blockIdx.x * blockDim.x + threadIdx.x;
    if (i >= 8000 * 256) return;
    int bi = i >> 8, d = i & 255;
    u[(long long)bi * 512 + d]       = seq[i] * ci0[bi];
    u[(long long)bi * 512 + 256 + d] = pg[i]  * ci1[bi];
}

__global__ void inter_sum_kernel(const float* __restrict__ inter, float* __restrict__ bs) {
    int b = blockIdx.x, tid = threadIdx.x;
    __shared__ float sb[256];
    const float* p = inter + (long long)b * 56000;
    float loc = 0.f;
    for (int i = tid; i < 56000; i += 256) loc += p[i];
    sb[tid] = loc; __syncthreads();
    for (int s = 128; s > 0; s >>= 1) {
        if (tid < s) sb[tid] += sb[tid + s];
        __syncthreads();
    }
    if (tid == 0) bs[b] = sb[0];
}

__global__ void inter_scale_kernel(float* __restrict__ inter, const float* __restrict__ bs) {
    int i = blockIdx.x * blockDim.x + threadIdx.x;
    if (i < 448000) inter[i] /= bs[i / 56000];
}

__global__ void cp_partial_kernel(const float* __restrict__ prot, const float* __restrict__ comp,
                                  const float* __restrict__ inter, float* __restrict__ part) {
    int chunk = blockIdx.x, b = blockIdx.y, d = threadIdx.x;
    int p0 = chunk * 32;
    int p1 = p0 + 32; if (p1 > 1000) p1 = 1000;
    const float* crow = comp + (long long)b * 56 * 256 + d;
    float acc = 0.f;
    for (int p = p0; p < p1; ++p) {
        float pv = prot[((long long)b * 1000 + p) * 256 + d];
        const float* irow = inter + ((long long)b * 1000 + p) * 56;
        for (int c = 0; c < 56; ++c)
            acc += irow[c] * tanhf(pv * crow[(long long)c * 256]);
    }
    part[((long long)b * 32 + chunk) * 256 + d] = acc;
}

__global__ void cp_reduce_kernel(const float* __restrict__ part, float* __restrict__ cp) {
    int b = blockIdx.x, d = threadIdx.x;
    float acc = 0.f;
    for (int c = 0; c < 32; ++c) acc += part[((long long)b * 32 + c) * 256 + d];
    cp[b * 256 + d] = acc;
}

__global__ void conv_pool_kernel(const float* __restrict__ cp, const float* __restrict__ W,
                                 const float* __restrict__ bia, float* __restrict__ out) {
    int b = blockIdx.x;
    for (int f = threadIdx.x; f < 2048; f += blockDim.x) {
        int o = f >> 5, q = f & 31;
        float mx = -3.4e38f;
        for (int s = 0; s < 4; ++s) {
            int t = q * 4 + s;
            float acc = bia[o];
            for (int k = 0; k < 4; ++k) {
                int pos = 2 * t - 1 + k;
                if (pos >= 0 && pos < 256) acc += cp[b * 256 + pos] * W[o * 4 + k];
            }
            float l = acc > 0.f ? acc : 0.1f * acc;
            mx = fmaxf(mx, l);
        }
        out[b * 2048 + f] = mx;
    }
}

// ------------------------------- host driver -------------------------------

static inline void launch_gemm(hipStream_t s,
    const float* A, const float* Bw, const float* bias, const float* addp, float* C,
    int M, int N, int K, int batch,
    long long sA, long long sB, long long sC,
    int lda, int ldb, int ldc, int transB, int act, int reluA)
{
    dim3 grid((N + 127) / 128, (M + 15) / 16, batch);
    wmma_gemm_kernel<<<grid, dim3(128, 1, 1), 0, s>>>(A, Bw, bias, addp, C, M, N, K,
                                                      sA, sB, sC, lda, ldb, ldc,
                                                      transB, act, reluA);
}

extern "C" void kernel_launch(void* const* d_in, const int* in_sizes, int n_in,
                              void* d_out, int out_size, void* d_ws, size_t ws_size,
                              hipStream_t stream) {
    (void)in_sizes; (void)n_in; (void)out_size; (void)ws_size;

    // ---- inputs (setup_inputs() dict insertion order, params flattened) ----
    const int*   prot_data = (const int*)  d_in[0];
    const float* drug_ver  = (const float*)d_in[1];
    const float* drug_adj  = (const float*)d_in[2];
    const float* contacts  = (const float*)d_in[3];
    const float* embed_t   = (const float*)d_in[4];
    const float* g0_Wih = (const float*)d_in[5],  *g0_Whh = (const float*)d_in[6];
    const float* g0_bih = (const float*)d_in[7],  *g0_bhh = (const float*)d_in[8];
    const float* g1_Wih = (const float*)d_in[9],  *g1_Whh = (const float*)d_in[10];
    const float* g1_bih = (const float*)d_in[11], *g1_bhh = (const float*)d_in[12];
    const float *attnW[7], *attnB[7], *l0W[7], *l0B[7], *l1W[7], *l1B[7];
    for (int l = 0; l < 7; ++l) {
        attnW[l] = (const float*)d_in[13 + 2 * l]; attnB[l] = (const float*)d_in[14 + 2 * l];
        l0W[l]   = (const float*)d_in[27 + 2 * l]; l0B[l]   = (const float*)d_in[28 + 2 * l];
        l1W[l]   = (const float*)d_in[41 + 2 * l]; l1B[l]   = (const float*)d_in[42 + 2 * l];
    }
    const float* gatFW = (const float*)d_in[55], *gatFB = (const float*)d_in[56];
    const float* c0W0 = (const float*)d_in[57], *c0B0 = (const float*)d_in[58];
    const float* c0W1 = (const float*)d_in[59], *c0B1 = (const float*)d_in[60];
    const float* c1W0 = (const float*)d_in[61], *c1B0 = (const float*)d_in[62];
    const float* c1W1 = (const float*)d_in[63], *c1B1 = (const float*)d_in[64];
    const float* ciOW = (const float*)d_in[65], *ciOB = (const float*)d_in[66];
    const float* gcW0 = (const float*)d_in[67], *gcB0 = (const float*)d_in[68];
    const float* gcW1 = (const float*)d_in[69], *gcB1 = (const float*)d_in[70];
    const float* gcW2 = (const float*)d_in[71], *gcB2 = (const float*)d_in[72];
    const float* gcFW = (const float*)d_in[73], *gcFB = (const float*)d_in[74];
    const float* jpW  = (const float*)d_in[75], *jpB  = (const float*)d_in[76];
    const float* jcW  = (const float*)d_in[77], *jcB  = (const float*)d_in[78];
    const float* cvW  = (const float*)d_in[79], *cvB  = (const float*)d_in[80];
    const float* r1W  = (const float*)d_in[81], *r1B  = (const float*)d_in[82];
    const float* r2W  = (const float*)d_in[83], *r2B  = (const float*)d_in[84];
    const float* r3W  = (const float*)d_in[85], *r3B  = (const float*)d_in[86];

    // ---- workspace layout (floats) ----
    float* ws     = (float*)d_ws;
    float* u_buf  = ws;                  // 8,000,000 : xp scratch / attn / concat
    float* emb_b  = u_buf  + 8000000;    // 2,048,000
    float* seqA_b = emb_b  + 2048000;    // 2,048,000 : GRU0 out
    float* seq_b  = seqA_b + 2048000;    // 2,048,000 : prot_seq
    float* xA_b   = seq_b  + 2048000;    // 2,048,000 : GAT ping / jp
    float* xB_b   = xA_b   + 2048000;    // 2,048,000 : GAT pong / mlp tmp
    float* t1_b   = xB_b   + 2048000;    // 2,048,000
    float* pg_b   = t1_b   + 2048000;    // 2,048,000 : prot_graph
    float* prot_b = pg_b   + 2048000;    // 2,048,000 : cross output
    float* sm     = prot_b + 2048000;
    float* h_b    = sm;                  // 81,920
    float* gh_b   = sm + 81920;          // 245,760
    float* cA_b   = sm + 327680;         // 114,688
    float* cB_b   = sm + 442368;         // 114,688
    float* jc_b   = sm + 557056;         // 114,688
    float* ci0_b  = sm + 671744;         // 8,000
    float* ci1_b  = sm + 679744;         // 8,000
    float* cp_b   = sm + 687744;         // 2,048
    float* cv_b   = sm + 689792;         // 16,384
    float* r1o_b  = sm + 706176;         // 4,800
    float* r2o_b  = sm + 710976;         // 2,400
    float* bs_b   = sm + 713376;         // 8
    float* part_b = sm + 713384;         // 65,536
    float* pver_b = sm + 778920;         // 21,504 : drug_ver padded to (8,56,48)
    float* pW0_b  = sm + 800424;         // 12,288 : gcW0 padded to (256,48)

    float* inter = (float*)d_out;            // (8,1000,56)
    float* affn  = (float*)d_out + 448000;   // (8,1)

    // ---- 1. embedding lookup ----
    embed_kernel<<<8000, 256, 0, stream>>>(prot_data, embed_t, emb_b, 8000);

    // ---- 2. GRU over J (320 seqs, T=25) ----
    launch_gemm(stream, emb_b, g0_Wih, g0_bih, nullptr, u_buf,
                8000, 768, 256, 1, 0, 0, 0, 256, 256, 768, 1, 0, 0);
    zero_kernel<<<(81920 + 255) / 256, 256, 0, stream>>>(h_b, 81920);
    for (int t = 0; t < 25; ++t) {
        launch_gemm(stream, h_b, g0_Whh, g0_bhh, nullptr, gh_b,
                    320, 768, 256, 1, 0, 0, 0, 256, 256, 768, 1, 0, 0);
        gru_step_kernel<<<(320 * 256 + 255) / 256, 256, 0, stream>>>(
            u_buf, gh_b, h_b, seqA_b, 320, 25, t);
    }

    // ---- 3. GRU over I (200 seqs, T=40) ----
    launch_gemm(stream, seqA_b, g1_Wih, g1_bih, nullptr, u_buf,
                8000, 768, 256, 1, 0, 0, 0, 256, 256, 768, 1, 0, 0);
    zero_kernel<<<(51200 + 255) / 256, 256, 0, stream>>>(h_b, 51200);
    for (int t = 0; t < 40; ++t) {
        launch_gemm(stream, h_b, g1_Whh, g1_bhh, nullptr, gh_b,
                    200, 768, 256, 1, 0, 0, 0, 256, 256, 768, 1, 0, 0);
        gru_step_kernel<<<(200 * 256 + 255) / 256, 256, 0, stream>>>(
            u_buf, gh_b, h_b, seq_b, 200, 40, t);
    }

    // ---- 4. GAT: 7 residual sigmoid-attention layers on emb ----
    float* bufs[2] = {xA_b, xB_b};
    const float* cur = emb_b;
    for (int l = 0; l < 7; ++l) {
        float* nxt = bufs[l & 1];
        launch_gemm(stream, cur, attnW[l], attnB[l], nullptr, t1_b,
                    1000, 256, 256, 8, 256000, 0, 256000, 256, 256, 256, 1, 0, 0);
        launch_gemm(stream, t1_b, cur, nullptr, nullptr, u_buf,
                    1000, 1000, 256, 8, 256000, 256000, 1000000, 256, 256, 1000, 1, 2, 0);
        gat_norm_kernel<<<dim3(1000, 8), 256, 0, stream>>>(u_buf, contacts);
        launch_gemm(stream, u_buf, cur, nullptr, nullptr, t1_b,
                    1000, 256, 1000, 8, 1000000, 256000, 256000, 1000, 256, 256, 0, 0, 0);
        launch_gemm(stream, t1_b, l0W[l], l0B[l], nullptr, u_buf,
                    1000, 256, 256, 8, 256000, 0, 256000, 256, 256, 256, 1, 1, 0);
        launch_gemm(stream, u_buf, l1W[l], l1B[l], cur, nxt,
                    1000, 256, 256, 8, 256000, 0, 256000, 256, 256, 256, 1, 1, 0);
        cur = nxt;
    }
    launch_gemm(stream, cur, gatFW, gatFB, nullptr, pg_b,
                1000, 256, 256, 8, 256000, 0, 256000, 256, 256, 256, 1, 0, 0);

    // ---- 5. cross interaction ----
    launch_gemm(stream, pg_b, c0W0, c0B0, nullptr, t1_b,
                1000, 256, 256, 8, 256000, 0, 256000, 256, 256, 256, 1, 1, 0);
    launch_gemm(stream, t1_b, c0W1, c0B1, nullptr, xB_b,
                1000, 256, 256, 8, 256000, 0, 256000, 256, 256, 256, 1, 0, 0);
    ci_dot_kernel<<<8000, 256, 0, stream>>>(xB_b, seq_b, ci0_b);
    launch_gemm(stream, seq_b, c1W0, c1B0, nullptr, t1_b,
                1000, 256, 256, 8, 256000, 0, 256000, 256, 256, 256, 1, 1, 0);
    launch_gemm(stream, t1_b, c1W1, c1B1, nullptr, xB_b,
                1000, 256, 256, 8, 256000, 0, 256000, 256, 256, 256, 1, 0, 0);
    ci_dot_kernel<<<8000, 256, 0, stream>>>(xB_b, pg_b, ci1_b);
    concat_kernel<<<8000, 256, 0, stream>>>(seq_b, pg_b, ci0_b, ci1_b, u_buf);
    launch_gemm(stream, u_buf, ciOW, ciOB, nullptr, prot_b,
                8000, 256, 512, 1, 0, 0, 0, 512, 512, 256, 1, 0, 0);

    // ---- 6. GCN on compound (layer-0 operands zero-padded K=43 -> 48) ----
    pad_copy_kernel<<<(448 * 48 + 255) / 256, 256, 0, stream>>>(drug_ver, pver_b, 448, 43, 48);
    pad_copy_kernel<<<(256 * 48 + 255) / 256, 256, 0, stream>>>(gcW0, pW0_b, 256, 43, 48);
    launch_gemm(stream, pver_b, pW0_b, gcB0, nullptr, cB_b,
                56, 256, 48, 8, 2688, 0, 14336, 48, 48, 256, 1, 0, 0);
    launch_gemm(stream, drug_adj, cB_b, nullptr, nullptr, cA_b,
                56, 256, 56, 8, 3136, 14336, 14336, 56, 256, 256, 0, 1, 0);
    launch_gemm(stream, cA_b, gcW1, gcB1, nullptr, cB_b,
                56, 256, 256, 8, 14336, 0, 14336, 256, 256, 256, 1, 0, 0);
    launch_gemm(stream, drug_adj, cB_b, nullptr, nullptr, cA_b,
                56, 256, 56, 8, 3136, 14336, 14336, 56, 256, 256, 0, 1, 0);
    launch_gemm(stream, cA_b, gcW2, gcB2, nullptr, cB_b,
                56, 256, 256, 8, 14336, 0, 14336, 256, 256, 256, 1, 0, 0);
    launch_gemm(stream, drug_adj, cB_b, nullptr, nullptr, cA_b,
                56, 256, 56, 8, 3136, 14336, 14336, 56, 256, 256, 0, 1, 0);
    launch_gemm(stream, cA_b, gcFW, gcFB, nullptr, cB_b,
                56, 256, 256, 8, 14336, 0, 14336, 256, 256, 256, 1, 0, 0);  // comp

    // ---- 7. jp/jc (fused input relu), inter, normalization ----
    launch_gemm(stream, prot_b, jpW, jpB, nullptr, xA_b,
                8000, 256, 256, 1, 0, 0, 0, 256, 256, 256, 1, 0, 1);
    launch_gemm(stream, cB_b, jcW, jcB, nullptr, jc_b,
                56, 256, 256, 8, 14336, 0, 14336, 256, 256, 256, 1, 0, 1);
    launch_gemm(stream, xA_b, jc_b, nullptr, nullptr, inter,
                1000, 56, 256, 8, 256000, 14336, 56000, 256, 256, 56, 1, 2, 0);
    inter_sum_kernel<<<8, 256, 0, stream>>>(inter, bs_b);
    inter_scale_kernel<<<(448000 + 255) / 256, 256, 0, stream>>>(inter, bs_b);

    // ---- 8. cp (deterministic two-stage reduction) ----
    cp_partial_kernel<<<dim3(32, 8), 256, 0, stream>>>(prot_b, cB_b, inter, part_b);
    cp_reduce_kernel<<<8, 256, 0, stream>>>(part_b, cp_b);

    // ---- 9. regressor ----
    conv_pool_kernel<<<8, 256, 0, stream>>>(cp_b, cvW, cvB, cv_b);
    launch_gemm(stream, cv_b, r1W, r1B, nullptr, r1o_b,
                8, 600, 2048, 1, 0, 0, 0, 2048, 2048, 600, 1, 3, 0);
    launch_gemm(stream, r1o_b, r2W, r2B, nullptr, r2o_b,
                8, 300, 600, 1, 0, 0, 0, 600, 600, 300, 1, 3, 0);
    launch_gemm(stream, r2o_b, r3W, r3B, nullptr, affn,
                8, 1, 300, 1, 0, 0, 0, 300, 300, 1, 1, 0, 0);
}
